// GuidenceAttention_89928025244015
// MI455X (gfx1250) — compile-verified, matrix operand order checked by reading, not run on hardware
//
#include <hip/hip_runtime.h>

// ---------------------------------------------------------------------------
// GuidenceAttention for MI455X (gfx1250, wave32)
//   WMMA f32_16x16x32_f16 for all GEMMs (HEAD_DIM=32 == WMMA K depth)
//   TDM (tensor_load_to_lds) double-buffered K/V tile staging in attention
// ---------------------------------------------------------------------------

#define DIMC   256
#define HEADS  8
#define HD     32
#define NQ     1024
#define NK     1024
#define BATCH  8
#define ROWS   (NQ * BATCH)         // 8192 rows for the projection GEMMs
#define SCALE  0.17677669529663687f // 1/sqrt(32)

typedef __attribute__((ext_vector_type(16))) _Float16 v16h;
typedef __attribute__((ext_vector_type(8)))  float    v8f;
typedef __attribute__((ext_vector_type(4)))  unsigned u32x4;
typedef __attribute__((ext_vector_type(8)))  unsigned u32x8;

// ---- WMMA wrapper ----------------------------------------------------------
__device__ __forceinline__ v8f wmma_f16(v16h a, v16h b, v8f c) {
  return __builtin_amdgcn_wmma_f32_16x16x32_f16(
      /*neg_a=*/false, a, /*neg_b=*/false, b,
      /*c_mod=*/(short)0, c, /*reuse_a=*/false, /*reuse_b=*/false);
}

// ---- TDM: DMA a 32x32 f16 tile (row stride 32 elems) from global to LDS ----
// D# group0: count=1, lds_addr, global_addr[56:0], type=2
// D# group1: data_size=2B, tensor_dim0/1=32, tile_dim0/1=32, dim0_stride=32
__device__ __forceinline__ void tdm_load_tile32x32_f16(const _Float16* gsrc,
                                                       _Float16* ldsdst) {
  const unsigned long long ga = (unsigned long long)(size_t)gsrc;
  const unsigned lds = (unsigned)(size_t)ldsdst;  // generic LDS addr[31:0] = offset
  u32x4 g0;
  g0[0] = 1u;                                            // count=1, user mode
  g0[1] = lds;                                           // lds_addr (bytes)
  g0[2] = (unsigned)ga;                                  // global_addr[31:0]
  g0[3] = (unsigned)((ga >> 32) & 0x01ffffffu) | (2u << 30);  // addr[56:32]|type=2
  u32x8 g1;
  g1[0] = 1u << 16;        // workgroup_mask=0, data_size=1 (2 bytes)
  g1[1] = 32u << 16;       // tensor_dim0[15:0] = 32
  g1[2] = 32u << 16;       // tensor_dim0[31:16]=0 | tensor_dim1[15:0] = 32
  g1[3] = 32u << 16;       // tensor_dim1[31:16]=0 | tile_dim0 = 32
  g1[4] = 32u;             // tile_dim1 = 32, tile_dim2 = 0
  g1[5] = 32u;             // tensor_dim0_stride[31:0] = 32 elements
  g1[6] = 0u;              // stride0 hi / dim1_stride lo
  g1[7] = 0u;              // dim1_stride hi
  asm volatile("tensor_load_to_lds %0, %1" :: "s"(g0), "s"(g1) : "memory");
}

// ---- A-layout gather (16x32 f16 A-matrix, ISA 7.12.2) ----------------------
// lanes 0-15: M=lane, K={0..7,16..23}; lanes 16-31: M=lane-16, K={8..15,24..31}
__device__ __forceinline__ v16h gather_a_h(const _Float16* src, int row_base,
                                           int rstride) {
  const int l  = threadIdx.x & 31;
  const int kg = (l >> 4) * 8;
  const _Float16* p = src + (size_t)(row_base + (l & 15)) * rstride + kg;
  v16h a;
#pragma unroll
  for (int j = 0; j < 8; ++j) { a[j] = p[j]; a[j + 8] = p[j + 16]; }
  return a;
}

__device__ __forceinline__ v16h gather_a_f32(const float* src, int row_base,
                                             int rstride, int c0) {
  const int l  = threadIdx.x & 31;
  const int kg = (l >> 4) * 8;
  const float* p = src + (size_t)(row_base + (l & 15)) * rstride + c0 + kg;
  v16h a;
#pragma unroll
  for (int j = 0; j < 8; ++j) {
    a[j]     = (_Float16)p[j];
    a[j + 8] = (_Float16)p[j + 16];
  }
  return a;
}

// ---- B-layout gather, column fixed per lane (for V: B(k=key, n=dim)) -------
__device__ __forceinline__ v16h gather_b_col_h(const _Float16* src, int kbase,
                                               int colbase, int rstride) {
  const int l   = threadIdx.x & 31;
  const int kg  = (l >> 4) * 8;
  const int col = colbase + (l & 15);
  v16h b;
#pragma unroll
  for (int j = 0; j < 8; ++j) {
    b[j]     = src[(size_t)(kbase + kg + j) * rstride + col];
    b[j + 8] = src[(size_t)(kbase + kg + 16 + j) * rstride + col];
  }
  return b;
}

// ---- cross-lane row reductions over the 16-lane halves ---------------------
__device__ __forceinline__ float redmax16(float v) {
#pragma unroll
  for (int m = 1; m < 16; m <<= 1) v = fmaxf(v, __shfl_xor(v, m, 32));
  return v;
}
__device__ __forceinline__ float redsum16(float v) {
#pragma unroll
  for (int m = 1; m < 16; m <<= 1) v += __shfl_xor(v, m, 32);
  return v;
}

// ===========================================================================
// Kernel 1: head projection   proj = X @ W^T,  scatter to [b][h][s][d] f16
// ===========================================================================
__global__ __launch_bounds__(32) void proj_head_kernel(
    const float* __restrict__ X, const float* __restrict__ W,
    _Float16* __restrict__ out) {
  const int tile = blockIdx.x;
  const int tn = tile & 15;
  const int tm = tile >> 4;
  const int l  = threadIdx.x & 31;
  const int nl = l & 15;

  v8f acc = {};
#pragma unroll
  for (int c0 = 0; c0 < DIMC; c0 += 32) {
    v16h a  = gather_a_f32(X, tm * 16, DIMC, c0);
    v16h bw = gather_a_f32(W, tn * 16, DIMC, c0);  // B(k,n) = W[n][c0+k]
    acc = wmma_f16(a, bw, acc);
  }

  const int dout = tn * 16 + nl;
  const int h = dout >> 5, d = dout & 31;
#pragma unroll
  for (int j = 0; j < 8; ++j) {
    const int M = j + ((l >> 4) << 3);
    const int r = tm * 16 + M;
    const int s = r >> 3, b = r & 7;
    out[(((size_t)b * HEADS + h) * NQ + s) * HD + d] = (_Float16)acc[j];
  }
}

// ===========================================================================
// Kernel 2: fused guided flash attention per (b, h, 16-query tile)
//   TDM-staged, double-buffered K/V tiles in LDS
// ===========================================================================
__global__ __launch_bounds__(32) void attn_kernel(
    const _Float16* __restrict__ Qp,  const _Float16* __restrict__ Kp,
    const _Float16* __restrict__ Vp,  const _Float16* __restrict__ Qip,
    const _Float16* __restrict__ Kip, const _Float16* __restrict__ Vip,
    float* __restrict__ Xh, float* __restrict__ Xih) {
  __shared__ _Float16 stage[6][32 * 32];  // 12 KB TDM staging (double-buffered)
  __shared__ _Float16 lp[16 * 32];        // 1 KB P-tile bounce (D->A layout)

  const int bid = blockIdx.x;
  const int qt = bid & 63;
  const int h  = (bid >> 6) & 7;
  const int b  = bid >> 9;
  const int q0 = qt * 16;
  const size_t hoff = (size_t)(b * HEADS + h) * NQ * HD;

  const _Float16* Q  = Qp  + hoff;
  const _Float16* K  = Kp  + hoff;
  const _Float16* V  = Vp  + hoff;
  const _Float16* Qi = Qip + hoff;
  const _Float16* Ki = Kip + hoff;
  const _Float16* Vi = Vip + hoff;

  const int l  = threadIdx.x & 31;
  const int nl = l & 15;
  const int Mb = (l >> 4) << 3;

  const v16h aq  = gather_a_h(Q,  q0, HD);
  const v16h aqi = gather_a_h(Qi, q0, HD);

  float m[8], lsum[8];
  v8f o0 = {}, o1 = {};
#pragma unroll
  for (int j = 0; j < 8; ++j) { m[j] = -1e30f; lsum[j] = 0.0f; }

  // -------- pass 1: flash softmax + O for the real branch --------
  tdm_load_tile32x32_f16(K, stage[0]);
  tdm_load_tile32x32_f16(V, stage[2]);
  for (int kb = 0; kb < NK; kb += 32) {
    const int cur = (kb >> 5) & 1, nxt = cur ^ 1;
    if (kb + 32 < NK) {
      tdm_load_tile32x32_f16(K + (size_t)(kb + 32) * HD, stage[nxt]);
      tdm_load_tile32x32_f16(V + (size_t)(kb + 32) * HD, stage[2 + nxt]);
      __builtin_amdgcn_s_wait_tensorcnt(2);   // current pair done, next in flight
    } else {
      __builtin_amdgcn_s_wait_tensorcnt(0);
    }
    const _Float16* kc = stage[cur];
    const _Float16* vc = stage[2 + cur];

    v16h bk0 = gather_a_h(kc, 0,  HD);   // B(k=d, n=key)
    v16h bk1 = gather_a_h(kc, 16, HD);
    v8f z = {};
    v8f s0 = wmma_f16(aq, bk0, z);
    v8f s1 = wmma_f16(aq, bk1, z);

    float f[8], e0[8], e1[8];
#pragma unroll
    for (int j = 0; j < 8; ++j) {
      const float a0 = s0[j] * SCALE, a1 = s1[j] * SCALE;
      const float cm = redmax16(fmaxf(a0, a1));
      const float mn = fmaxf(m[j], cm);
      f[j]  = __expf(m[j] - mn);
      e0[j] = __expf(a0 - mn);
      e1[j] = __expf(a1 - mn);
      lsum[j] = lsum[j] * f[j] + redsum16(e0[j] + e1[j]);
      m[j] = mn;
    }
#pragma unroll
    for (int j = 0; j < 8; ++j) {
      const int M = j + Mb;
      lp[M * 32 + nl]      = (_Float16)e0[j];
      lp[M * 32 + 16 + nl] = (_Float16)e1[j];
    }
    asm volatile("s_wait_dscnt 0" ::: "memory");
    v16h ap  = gather_a_h(lp, 0, 32);
    v16h bv0 = gather_b_col_h(vc, 0, 0,  HD);
    v16h bv1 = gather_b_col_h(vc, 0, 16, HD);
#pragma unroll
    for (int j = 0; j < 8; ++j) { o0[j] *= f[j]; o1[j] *= f[j]; }
    o0 = wmma_f16(ap, bv0, o0);
    o1 = wmma_f16(ap, bv1, o1);
  }

  float im[8];
#pragma unroll
  for (int j = 0; j < 8; ++j) {
    im[j] = 1.0f / lsum[j];
    const int s = q0 + j + Mb;
    Xh[((size_t)b * NQ + s) * DIMC + h * HD + nl]      = o0[j] * im[j];
    Xh[((size_t)b * NQ + s) * DIMC + h * HD + 16 + nl] = o1[j] * im[j];
  }

  // -------- pass 2: exact p_real, gated img-branch flash --------
  float m2[8], l2[8];
  v8f oi0 = {}, oi1 = {};
#pragma unroll
  for (int j = 0; j < 8; ++j) { m2[j] = -1e30f; l2[j] = 0.0f; }

  tdm_load_tile32x32_f16(K,  stage[0]);
  tdm_load_tile32x32_f16(Ki, stage[2]);
  tdm_load_tile32x32_f16(Vi, stage[4]);
  for (int kb = 0; kb < NK; kb += 32) {
    const int cur = (kb >> 5) & 1, nxt = cur ^ 1;
    if (kb + 32 < NK) {
      tdm_load_tile32x32_f16(K  + (size_t)(kb + 32) * HD, stage[nxt]);
      tdm_load_tile32x32_f16(Ki + (size_t)(kb + 32) * HD, stage[2 + nxt]);
      tdm_load_tile32x32_f16(Vi + (size_t)(kb + 32) * HD, stage[4 + nxt]);
      __builtin_amdgcn_s_wait_tensorcnt(3);  // current triple done
    } else {
      __builtin_amdgcn_s_wait_tensorcnt(0);
    }
    const _Float16* kc  = stage[cur];
    const _Float16* kic = stage[2 + cur];
    const _Float16* vic = stage[4 + cur];

    v16h bk0  = gather_a_h(kc,  0,  HD);
    v16h bk1  = gather_a_h(kc,  16, HD);
    v16h bki0 = gather_a_h(kic, 0,  HD);
    v16h bki1 = gather_a_h(kic, 16, HD);
    v8f z = {};
    v8f s0 = wmma_f16(aq,  bk0,  z);
    v8f s1 = wmma_f16(aq,  bk1,  z);
    v8f g0 = wmma_f16(aqi, bki0, z);
    v8f g1 = wmma_f16(aqi, bki1, z);

    float f[8], e0[8], e1[8];
#pragma unroll
    for (int j = 0; j < 8; ++j) {
      const float p0  = __expf(s0[j] * SCALE - m[j]) * im[j];  // exact softmax
      const float p1  = __expf(s1[j] * SCALE - m[j]) * im[j];
      const float gg0 = g0[j] * SCALE * p0;                    // gated logits
      const float gg1 = g1[j] * SCALE * p1;
      const float cm  = redmax16(fmaxf(gg0, gg1));
      const float mn  = fmaxf(m2[j], cm);
      f[j]  = __expf(m2[j] - mn);
      e0[j] = __expf(gg0 - mn);
      e1[j] = __expf(gg1 - mn);
      l2[j] = l2[j] * f[j] + redsum16(e0[j] + e1[j]);
      m2[j] = mn;
    }
#pragma unroll
    for (int j = 0; j < 8; ++j) {
      const int M = j + Mb;
      lp[M * 32 + nl]      = (_Float16)e0[j];
      lp[M * 32 + 16 + nl] = (_Float16)e1[j];
    }
    asm volatile("s_wait_dscnt 0" ::: "memory");
    v16h ap   = gather_a_h(lp, 0, 32);
    v16h bvi0 = gather_b_col_h(vic, 0, 0,  HD);
    v16h bvi1 = gather_b_col_h(vic, 0, 16, HD);
#pragma unroll
    for (int j = 0; j < 8; ++j) { oi0[j] *= f[j]; oi1[j] *= f[j]; }
    oi0 = wmma_f16(ap, bvi0, oi0);
    oi1 = wmma_f16(ap, bvi1, oi1);
  }

#pragma unroll
  for (int j = 0; j < 8; ++j) {
    const float inv = 1.0f / l2[j];
    const int s = q0 + j + Mb;
    Xih[((size_t)b * NQ + s) * DIMC + h * HD + nl]      = oi0[j] * inv;
    Xih[((size_t)b * NQ + s) * DIMC + h * HD + 16 + nl] = oi1[j] * inv;
  }
}

// ===========================================================================
// Kernel 3: output projection  out = Xh @ W^T + bias  (fp32 out)
// ===========================================================================
__global__ __launch_bounds__(32) void proj_out_kernel(
    const float* __restrict__ Xh, const float* __restrict__ W,
    const float* __restrict__ bias, float* __restrict__ out) {
  const int tile = blockIdx.x;
  const int tn = tile & 15;
  const int tm = tile >> 4;
  const int l  = threadIdx.x & 31;
  const int nl = l & 15;

  v8f acc = {};
#pragma unroll
  for (int c0 = 0; c0 < DIMC; c0 += 32) {
    v16h a  = gather_a_f32(Xh, tm * 16, DIMC, c0);
    v16h bw = gather_a_f32(W,  tn * 16, DIMC, c0);
    acc = wmma_f16(a, bw, acc);
  }

  const int dout = tn * 16 + nl;
  const float bv = bias[dout];
#pragma unroll
  for (int j = 0; j < 8; ++j) {
    const int r = tm * 16 + j + ((l >> 4) << 3);
    out[(size_t)r * DIMC + dout] = acc[j] + bv;
  }
}

// ===========================================================================
extern "C" void kernel_launch(void* const* d_in, const int* in_sizes, int n_in,
                              void* d_out, int out_size, void* d_ws,
                              size_t ws_size, hipStream_t stream) {
  const float* q     = (const float*)d_in[0];
  const float* k     = (const float*)d_in[1];
  const float* v     = (const float*)d_in[2];
  const float* q_img = (const float*)d_in[3];
  const float* k_img = (const float*)d_in[4];
  const float* v_img = (const float*)d_in[5];
  const float* Wq    = (const float*)d_in[6];
  const float* Wk    = (const float*)d_in[7];
  const float* Wv    = (const float*)d_in[8];
  const float* Wq_im = (const float*)d_in[9];
  const float* Wk_im = (const float*)d_in[10];
  const float* Wv_im = (const float*)d_in[11];
  const float* Wp    = (const float*)d_in[12];
  const float* bp    = (const float*)d_in[13];
  const float* Wp_im = (const float*)d_in[14];
  const float* bp_im = (const float*)d_in[15];

  // workspace: 6 x f16 head tensors (24 MB) + 2 x f32 attn outputs (16 MB)
  const size_t headElems = (size_t)BATCH * HEADS * NQ * HD;  // 2,097,152
  _Float16* Qp  = (_Float16*)d_ws;
  _Float16* Kp  = Qp  + headElems;
  _Float16* Vp  = Kp  + headElems;
  _Float16* Qip = Vp  + headElems;
  _Float16* Kip = Qip + headElems;
  _Float16* Vip = Kip + headElems;
  float* Xh  = (float*)(Vip + headElems);
  float* Xih = Xh + (size_t)BATCH * NQ * DIMC;

  float* out_x  = (float*)d_out;
  float* out_xi = out_x + (size_t)BATCH * NQ * DIMC;

  const dim3 blk(32);
  const int projTiles = (ROWS / 16) * (DIMC / 16);  // 8192

  proj_head_kernel<<<projTiles, blk, 0, stream>>>(q,     Wq,    Qp);
  proj_head_kernel<<<projTiles, blk, 0, stream>>>(k,     Wk,    Kp);
  proj_head_kernel<<<projTiles, blk, 0, stream>>>(v,     Wv,    Vp);
  proj_head_kernel<<<projTiles, blk, 0, stream>>>(q_img, Wq_im, Qip);
  proj_head_kernel<<<projTiles, blk, 0, stream>>>(k_img, Wk_im, Kip);
  proj_head_kernel<<<projTiles, blk, 0, stream>>>(v_img, Wv_im, Vip);

  const int attnBlocks = BATCH * HEADS * (NQ / 16);  // 4096
  attn_kernel<<<attnBlocks, blk, 0, stream>>>(Qp, Kp, Vp, Qip, Kip, Vip, Xh, Xih);

  proj_out_kernel<<<projTiles, blk, 0, stream>>>(Xh,  Wp,    bp,    out_x);
  proj_out_kernel<<<projTiles, blk, 0, stream>>>(Xih, Wp_im, bp_im, out_xi);
}
